// GCN_34282428957021
// MI455X (gfx1250) — matrix operand order repaired and checked by reading, not compiled, not run on hardware
//
#include <hip/hip_runtime.h>
#include <hip/hip_bf16.h>
#include <stdint.h>

// ---------------------------------------------------------------------------
// GCN (3x GCNConv + ReLU + dropout) for MI455X / gfx1250.
// - Dense GEMMs: v_wmma_f32_16x16x32_f16; one wave owns a 16-row strip and
//   sweeps all N-tiles so A fragments are loaded once (global b128) and B
//   fragments come from LDS (ds b128).
// - Weight staging into LDS uses the Tensor Data Mover (tensor_load_to_lds,
//   TENSORcnt) issued by wave 0.  (6-arg builtin variant on this toolchain.)
// - Edge aggregation (bandwidth-bound part): float4 gathers + global f32
//   atomics + global_prefetch_b8.
// ---------------------------------------------------------------------------

typedef __attribute__((ext_vector_type(16))) _Float16 v16h;
typedef __attribute__((ext_vector_type(8)))  _Float16 v8h;
typedef __attribute__((ext_vector_type(8)))  float    v8f;
typedef __attribute__((ext_vector_type(4)))  unsigned int v4u;
typedef __attribute__((ext_vector_type(8)))  int      v8i;
typedef __attribute__((ext_vector_type(4)))  int      v4i;

#define IN_DIM 128
#define D1 64
#define D2 32
#define D3 16

// ---------------- degree / normalization ----------------

__global__ void k_init_deg(float* __restrict__ deg, int n) {
    int i = blockIdx.x * blockDim.x + threadIdx.x;
    if (i < n) deg[i] = 1.0f;   // self-loop
}

__global__ void k_deg_accum(const long long* __restrict__ dst,
                            float* __restrict__ deg, int e) {
    int i = blockIdx.x * blockDim.x + threadIdx.x;
    if (i < e) atomicAdd(&deg[(int)dst[i]], 1.0f);
}

__global__ void k_dinv(float* __restrict__ d, int n) {
    int i = blockIdx.x * blockDim.x + threadIdx.x;
    if (i < n) {
        float v = d[i];
        d[i] = (v > 0.0f) ? rsqrtf(v) : 0.0f;
    }
}

// ---------------- f32 -> f16 conversions ----------------

__global__ void k_f32_to_f16(const float* __restrict__ in,
                             _Float16* __restrict__ out, long n) {
    long i = (long)blockIdx.x * blockDim.x + threadIdx.x;
    if (i < n) out[i] = (_Float16)in[i];
}

// W is [K, N] row-major (fan_in, fan_out); produce Wt [N, K] f16 so each
// B-matrix fragment is contiguous along K.
__global__ void k_w_transpose_f16(const float* __restrict__ W,
                                  _Float16* __restrict__ Wt, int K, int N) {
    int i = blockIdx.x * blockDim.x + threadIdx.x;
    if (i < K * N) {
        int c = i / K, k = i % K;
        Wt[i] = (_Float16)W[k * N + c];
    }
}

// ---------------- WMMA GEMM: Out[M,NOUT] = A[M,K](f16) * W[K,NOUT] ----------
// One wave per 16-row strip of A; A fragments resident in VGPRs, swept
// across all N-tiles with B fragments from LDS (staged by the TDM).

template <int K, int NOUT>
__global__ __launch_bounds__(256) void k_gemm_wmma(
    const _Float16* __restrict__ A, const _Float16* __restrict__ Wt,
    float* __restrict__ Out, int M) {
    constexpr unsigned T = (unsigned)K * NOUT;   // f16 elements of W
    static_assert(T <= 65535u, "tile_dim0 is a 16-bit field");
    __shared__ _Float16 ldsW[K * NOUT];

    // --- TDM: stage Wt (T f16 elements, contiguous) into LDS; wave 0 issues.
    if (threadIdx.x < 32) {
        unsigned lds_off = (unsigned)(size_t)(&ldsW[0]);
        unsigned long long ga = (unsigned long long)(size_t)Wt;
        // D# group0: count=1 | lds_addr | global_addr[56:0] | type=2
        v4u g0 = { 1u,
                   lds_off,
                   (unsigned)ga,
                   (unsigned)((ga >> 32) & 0x01FFFFFFu) | 0x80000000u };
        // D# group1: data_size=2B; tensor_dim0=T; tensor_dim1=1;
        //            tile_dim0=T; tile_dim1=1; tensor_dim0_stride=T.
        v8i g1 = { (int)0x00010000u,
                   (int)((T & 0xFFFFu) << 16),
                   (int)(((T >> 16) & 0xFFFFu) | (1u << 16)),
                   (int)((T & 0xFFFFu) << 16),
                   1,
                   (int)T,
                   0, 0 };
        v4i gz4 = { 0, 0, 0, 0 };
        v8i gz8 = { 0, 0, 0, 0, 0, 0, 0, 0 };
        __builtin_amdgcn_tensor_load_to_lds(g0, g1, gz4, gz4, gz8, 0);
        __builtin_amdgcn_s_wait_tensorcnt(0);
    }
    __syncthreads();

    const int waveId = blockIdx.x * (blockDim.x >> 5) + (threadIdx.x >> 5);
    const int mTiles = M / 16;
    if (waveId >= mTiles) return;   // wave-uniform: EXEC stays all-ones

    const int lane = threadIdx.x & 31;
    const int half = lane >> 4;
    const int idx  = lane & 15;
    const int row  = waveId * 16 + idx;

    // Load all A fragments for this strip once (resident across N-tiles).
    constexpr int KS = K / 32;
    v16h a[KS];
#pragma unroll
    for (int ks = 0; ks < KS; ++ks) {
        // A (16x32 f16): lanes<16 hold K {kb..kb+7, kb+16..kb+23},
        //                lanes>=16 hold K {kb+8..kb+15, kb+24..kb+31}.
        const _Float16* ap = A + (size_t)row * K + ks * 32 + half * 8;
        v8h alo = *(const v8h*)(ap);
        v8h ahi = *(const v8h*)(ap + 16);
#pragma unroll
        for (int i = 0; i < 8; ++i) { a[ks][i] = alo[i]; a[ks][i + 8] = ahi[i]; }
    }

#pragma unroll
    for (int nt = 0; nt < NOUT / 16; ++nt) {
        const int col = nt * 16 + idx;
        v8f acc = {};
#pragma unroll
        for (int ks = 0; ks < KS; ++ks) {
            // B (32x16 f16): lanes<16 hold K {kb..kb+7, kb+16..kb+23},
            //                lanes>=16 hold K {kb+8..kb+15, kb+24..kb+31}.
            const _Float16* bp = ldsW + (size_t)col * K + ks * 32 + half * 8;
            v8h blo = *(const v8h*)(bp);
            v8h bhi = *(const v8h*)(bp + 16);
            v16h b;
#pragma unroll
            for (int i = 0; i < 8; ++i) { b[i] = blo[i]; b[i + 8] = bhi[i]; }
            acc = __builtin_amdgcn_wmma_f32_16x16x32_f16(
                false, a[ks], false, b, (short)0, acc, false, false);
        }
        // D layout: VGPR r -> M = r + half*8, N = idx.
        float* op = Out + (size_t)(waveId * 16 + half * 8) * NOUT + nt * 16 + idx;
#pragma unroll
        for (int r = 0; r < 8; ++r) op[(size_t)r * NOUT] = acc[r];
    }
}

// ---------------- aggregation (self-loop init + edge scatter) ---------------

template <int C>
__global__ void k_agg_init(const float* __restrict__ t,
                           const float* __restrict__ dinv,
                           const float* __restrict__ bias,
                           float* __restrict__ out, int n) {
    long i = (long)blockIdx.x * blockDim.x + threadIdx.x;
    if (i >= (long)n * C) return;
    int node = (int)(i / C);
    int c = (int)(i % C);
    float di = dinv[node];
    out[i] = bias[c] + di * di * t[i];
}

template <int C>
__global__ void k_agg_edges(const long long* __restrict__ src,
                            const long long* __restrict__ dst,
                            const float* __restrict__ dinv,
                            const float* __restrict__ t,
                            float* __restrict__ out, int e) {
    constexpr int G = C / 4;
    long tid = (long)blockIdx.x * blockDim.x + threadIdx.x;
    if (tid >= (long)e * G) return;
    int ed = (int)(tid / G);
    int cg = (int)(tid % G) * 4;
    int s = (int)src[ed];
    int d = (int)dst[ed];
    float coef = dinv[s] * dinv[d];
    __builtin_prefetch(out + (size_t)d * C + cg, 1, 0);   // global_prefetch_b8
    float4 v = *(const float4*)(t + (size_t)s * C + cg);
    float* o = out + (size_t)d * C + cg;
    atomicAdd(o + 0, coef * v.x);
    atomicAdd(o + 1, coef * v.y);
    atomicAdd(o + 2, coef * v.z);
    atomicAdd(o + 3, coef * v.w);
}

// ---------------- ReLU + dropout -> f16 -------------------------------------

__global__ void k_relu_dropout_f16(const float* __restrict__ in,
                                   _Float16* __restrict__ out, long n,
                                   uint32_t seed) {
    long i = (long)blockIdx.x * blockDim.x + threadIdx.x;
    if (i >= n) return;
    float v = in[i];
    v = v > 0.0f ? v : 0.0f;
    uint32_t r = (uint32_t)i ^ seed;
    r *= 0x9E3779B9u; r ^= r >> 16;
    r *= 0x85EBCA6Bu; r ^= r >> 13;
    r *= 0xC2B2AE35u; r ^= r >> 16;
    // keep prob 0.8, scale by 1/0.8
    v = ((r >> 8) < 13421773u) ? v * 1.25f : 0.0f;
    out[i] = (_Float16)v;
}

// ---------------- launch ----------------------------------------------------

extern "C" void kernel_launch(void* const* d_in, const int* in_sizes, int n_in,
                              void* d_out, int out_size, void* d_ws, size_t ws_size,
                              hipStream_t stream) {
    const float* x  = (const float*)d_in[0];
    const long long* ei = (const long long*)d_in[1];   // int64 [2, E]
    const float* W1 = (const float*)d_in[2];
    const float* b1 = (const float*)d_in[3];
    const float* W2 = (const float*)d_in[4];
    const float* b2 = (const float*)d_in[5];
    const float* W3 = (const float*)d_in[6];
    const float* b3 = (const float*)d_in[7];
    float* out = (float*)d_out;

    const int N = in_sizes[0] / IN_DIM;
    const int E = in_sizes[1] / 2;
    const long long* src = ei;
    const long long* dst = ei + E;

    // Carve workspace (256B-aligned slabs).
    char* ws = (char*)d_ws;
    size_t off = 0;
    auto take = [&](size_t bytes) -> void* {
        void* p = (void*)(ws + off);
        off += (bytes + 255) & ~(size_t)255;
        return p;
    };
    float*    dinv = (float*)take((size_t)N * 4);
    _Float16* xh   = (_Float16*)take((size_t)N * IN_DIM * 2);  // reused as h1h/h2h
    _Float16* w1t  = (_Float16*)take((size_t)IN_DIM * D1 * 2);
    _Float16* w2t  = (_Float16*)take((size_t)D1 * D2 * 2);
    _Float16* w3t  = (_Float16*)take((size_t)D2 * D3 * 2);
    float*    bufA = (float*)take((size_t)N * D1 * 4);          // GEMM outputs
    float*    bufB = (float*)take((size_t)N * D1 * 4);          // aggregates

    const int T = 256;
    auto cdiv = [](long a, long b) { return (int)((a + b - 1) / b); };

    // Normalization.
    k_init_deg<<<cdiv(N, T), T, 0, stream>>>(dinv, N);
    k_deg_accum<<<cdiv(E, T), T, 0, stream>>>(dst, dinv, E);
    k_dinv<<<cdiv(N, T), T, 0, stream>>>(dinv, N);

    // Precision conversions.
    k_f32_to_f16<<<cdiv((long)N * IN_DIM, T), T, 0, stream>>>(x, xh, (long)N * IN_DIM);
    k_w_transpose_f16<<<cdiv(IN_DIM * D1, T), T, 0, stream>>>(W1, w1t, IN_DIM, D1);
    k_w_transpose_f16<<<cdiv(D1 * D2, T), T, 0, stream>>>(W2, w2t, D1, D2);
    k_w_transpose_f16<<<cdiv(D2 * D3, T), T, 0, stream>>>(W3, w3t, D2, D3);

    // ---- Layer 1: h1 = x @ W1 ; aggregate ; relu+dropout -> f16 ----
    {
        int mTiles = N / 16;
        k_gemm_wmma<IN_DIM, D1><<<cdiv(mTiles, 8), T, 0, stream>>>(xh, w1t, bufA, N);
        k_agg_init<D1><<<cdiv((long)N * D1, T), T, 0, stream>>>(bufA, dinv, b1, bufB, N);
        k_agg_edges<D1><<<cdiv((long)E * (D1 / 4), T), T, 0, stream>>>(src, dst, dinv, bufA, bufB, E);
        k_relu_dropout_f16<<<cdiv((long)N * D1, T), T, 0, stream>>>(bufB, xh, (long)N * D1, 0x2A0001u);
    }

    // ---- Layer 2 ----
    {
        int mTiles = N / 16;
        k_gemm_wmma<D1, D2><<<cdiv(mTiles, 8), T, 0, stream>>>(xh, w2t, bufA, N);
        k_agg_init<D2><<<cdiv((long)N * D2, T), T, 0, stream>>>(bufA, dinv, b2, bufB, N);
        k_agg_edges<D2><<<cdiv((long)E * (D2 / 4), T), T, 0, stream>>>(src, dst, dinv, bufA, bufB, E);
        k_relu_dropout_f16<<<cdiv((long)N * D2, T), T, 0, stream>>>(bufB, xh, (long)N * D2, 0x2A0002u);
    }

    // ---- Layer 3: output straight into d_out ----
    {
        int mTiles = N / 16;
        k_gemm_wmma<D2, D3><<<cdiv(mTiles, 8), T, 0, stream>>>(xh, w3t, bufA, N);
        k_agg_init<D3><<<cdiv((long)N * D3, T), T, 0, stream>>>(bufA, dinv, b3, out, N);
        k_agg_edges<D3><<<cdiv((long)E * (D3 / 4), T), T, 0, stream>>>(src, dst, dinv, bufA, out, E);
    }
}
